// LinearAttention_73383811220087
// MI455X (gfx1250) — matrix operand order repaired and checked by reading, not compiled, not run on hardware
//
#include <hip/hip_runtime.h>
#include <hip/hip_bf16.h>

// CDNA5 (gfx1250) chunked causal linear attention, bf16 WMMA, fp32 accum.
// [n=8, l=2048, h=16, d=64, m=64]; grid = n*h*MSPLIT blocks of 128 threads.
// Second-stage GEMMs computed transposed (out^T = V^T A^T + S^T Q^T,
// S^T += V^T K) so every WMMA operand is a contiguous LDS access.

typedef __attribute__((ext_vector_type(16))) __bf16 v16bf;
typedef __attribute__((ext_vector_type(8)))  __bf16 v8bf;
typedef __attribute__((ext_vector_type(8)))  float  v8f;

#define SEQ     2048
#define HEADS   16
#define DIM     64
#define CHUNK   64
#define NCHUNK  (SEQ / CHUNK)
#define MSPLIT  2
#define MCOLS   (DIM / MSPLIT)   // 32 value columns per block
#define MTILES  (MCOLS / 16)     // 2 WMMA tiles along value columns
#define LDW     72               // padded bf16 row stride (all LDS tiles)

__device__ __forceinline__ float phi(float x) {
    // elu(x) + 1
    return x > 0.0f ? (x + 1.0f) : __expf(x);
}

__device__ __forceinline__ v8f wmma_bf16(v16bf a, v16bf b, v8f c) {
    // D = A(16x32) * B(32x16) + C, fp32 accumulate
    return __builtin_amdgcn_wmma_f32_16x16x32_bf16(
        /*neg_a=*/false, a, /*neg_b=*/false, b,
        /*c_mod=*/(short)0, c, /*reuse_a=*/false, /*reuse_b=*/false);
}

// ---- wave32 fragment loaders (k = 16*(L/16) + j, same order in A and B), all contiguous in j ----
// A 16x32 (MxK): lane m = L%16, kbase = (L>=16)*16;
//   half j: K = kbase + j
__device__ __forceinline__ v16bf frag_a(const __bf16* s, int ld, int row0, int k0) {
    const int lane = threadIdx.x & 31;
    const int m  = lane & 15;
    const int kb = (lane >> 4) * 16;
    v16bf a;
#pragma unroll
    for (int j = 0; j < 16; ++j) {
        int k = kb + j;
        a[j] = s[(row0 + m) * ld + k0 + k];
    }
    return a;
}
// B 32x16 (KxN): lane n = L%16, kbase = (L>=16)*16; half j: K = kbase + j
__device__ __forceinline__ v16bf frag_b(const __bf16* s, int ld, int k0, int n0) {
    const int lane = threadIdx.x & 31;
    const int n  = lane & 15;
    const int kb = (lane >> 4) << 4;
    v16bf b;
#pragma unroll
    for (int j = 0; j < 16; ++j)
        b[j] = s[(k0 + kb + j) * ld + n0 + n];
    return b;
}
// B transposed source: B(k,n) = s[(n0+n)*ld + k0+k]
__device__ __forceinline__ v16bf frag_bT(const __bf16* s, int ld, int k0, int n0) {
    const int lane = threadIdx.x & 31;
    const int n  = lane & 15;
    const int kb = (lane >> 4) << 4;
    v16bf b;
#pragma unroll
    for (int j = 0; j < 16; ++j)
        b[j] = s[(n0 + n) * ld + k0 + kb + j];
    return b;
}
// B with every column equal to vec[k] (contiguous LDS broadcast):
// D(m,n) = sum_k A(m,k)*vec[k] for all n -> per-row dot product in every lane.
__device__ __forceinline__ v16bf frag_b_bcast(const __bf16* vec, int k0) {
    const int lane = threadIdx.x & 31;
    const int kb = (lane >> 4) << 4;
    v16bf b;
#pragma unroll
    for (int j = 0; j < 16; ++j)
        b[j] = vec[k0 + kb + j];
    return b;
}

__global__ __launch_bounds__(128)
void LinearAttention_73383811220087_kernel(const float* __restrict__ q,
                                           const float* __restrict__ k,
                                           const float* __restrict__ v,
                                           float* __restrict__ out) {
    __shared__ __align__(16) __bf16 Qs [CHUNK * LDW];  // phi(q), row-major [t][d]
    __shared__ __align__(16) __bf16 Ks [CHUNK * LDW];  // phi(k), row-major [t][d]
    __shared__ __align__(16) __bf16 As [CHUNK * LDW];  // masked scores, col-major [t_col][t_row]
    __shared__ __align__(16) __bf16 VsT[MCOLS * LDW];  // v transposed [vcol][t]
    __shared__ __align__(16) __bf16 Ss [MCOLS * LDW];  // S^T_prev [vcol][d]
    __shared__ float  ksum[DIM];        // running column-sum of phi(k), fp32
    __shared__ __align__(16) __bf16 ksbf[DIM];         // per-chunk bf16 copy
    __shared__ float  zint[CHUNK];      // full normalizer (intra + inter)

    const int tid  = threadIdx.x;
    const int lane = tid & 31;
    const int wave = tid >> 5;          // 4 waves: 16-row time strips / 16-col d strips
    const int n_lo = lane & 15;
    const int hi8  = (lane >> 4) * 8;

    const int bh = blockIdx.x >> 1;     // (batch, head)
    const int mh = blockIdx.x & 1;      // value-column half
    const int b  = bh >> 4;
    const int h  = bh & 15;
    const int mo = mh * MCOLS;

    const size_t base = ((size_t)b * SEQ * HEADS + h) * DIM;
    const float* qg = q + base;
    const float* kg = k + base;
    const float* vg = v + base;
    float*       og = out + base;
    const int rowstride = HEADS * DIM;  // 1024 floats per timestep

    if (tid < DIM) ksum[tid] = 0.0f;

    // Constant all-ones A fragment: 1^T * K gives column sums on the WMMA unit.
    v16bf ones;
#pragma unroll
    for (int j = 0; j < 16; ++j) ones[j] = (__bf16)1.0f;

    // Running state S^T (vcol-tile mt rows [mt*16,+16) x d-cols [wave*16,+16)).
    v8f Sacc[MTILES];
#pragma unroll
    for (int t = 0; t < MTILES; ++t)
#pragma unroll
        for (int e = 0; e < 8; ++e) Sacc[t][e] = 0.0f;

    for (int c = 0; c < NCHUNK; ++c) {
        const int t0 = c * CHUNK;

        // ---- stage S^T_prev and ksum (bf16) as WMMA operands ----
        // Sacc tile mt: lane holds (vcol = mt*16 + r + hi8, d = wave*16 + n_lo)
#pragma unroll
        for (int mt = 0; mt < MTILES; ++mt)
#pragma unroll
            for (int r = 0; r < 8; ++r)
                Ss[(mt * 16 + r + hi8) * LDW + wave * 16 + n_lo] = (__bf16)Sacc[mt][r];
        if (tid < DIM) ksbf[tid] = (__bf16)ksum[tid];

        // ---- cooperative global -> LDS, feature map fused, float4 loads ----
        for (int i = tid; i < CHUNK * 16; i += 128) {
            const int row = i >> 4;
            const int c4  = (i & 15) << 2;
            const size_t g = (size_t)(t0 + row) * rowstride + c4;
            float4 qv = *(const float4*)(qg + g);
            float4 kv = *(const float4*)(kg + g);
            __bf16* qd = &Qs[row * LDW + c4];
            __bf16* kd = &Ks[row * LDW + c4];
            qd[0] = (__bf16)phi(qv.x); qd[1] = (__bf16)phi(qv.y);
            qd[2] = (__bf16)phi(qv.z); qd[3] = (__bf16)phi(qv.w);
            kd[0] = (__bf16)phi(kv.x); kd[1] = (__bf16)phi(kv.y);
            kd[2] = (__bf16)phi(kv.z); kd[3] = (__bf16)phi(kv.w);
            if (c + 1 < NCHUNK) {
                __builtin_prefetch(qg + g + (size_t)CHUNK * rowstride, 0, 0);
                __builtin_prefetch(kg + g + (size_t)CHUNK * rowstride, 0, 0);
            }
        }
        for (int i = tid; i < CHUNK * (MCOLS / 4); i += 128) {
            const int row = i >> 3;
            const int c4  = (i & 7) << 2;
            const size_t g = (size_t)(t0 + row) * rowstride + mo + c4;
            float4 vv = *(const float4*)(vg + g);
            // store transposed: VsT[vcol][t]
            VsT[(c4 + 0) * LDW + row] = (__bf16)vv.x;
            VsT[(c4 + 1) * LDW + row] = (__bf16)vv.y;
            VsT[(c4 + 2) * LDW + row] = (__bf16)vv.z;
            VsT[(c4 + 3) * LDW + row] = (__bf16)vv.w;
            if (c + 1 < NCHUNK)
                __builtin_prefetch(vg + g + (size_t)CHUNK * rowstride, 0, 0);
        }
        __syncthreads();

        // ---- scores = phi(Q)_strip * phi(K)^T  (16x64, 4 col tiles) ----
        const v16bf aq0 = frag_a(Qs, LDW, wave * 16, 0);
        const v16bf aq1 = frag_a(Qs, LDW, wave * 16, 32);
        v8f sc[4];
#pragma unroll
        for (int nt = 0; nt < 4; ++nt) {
            v8f acc = {};
            acc = wmma_bf16(aq0, frag_bT(Ks, LDW, 0,  nt * 16), acc);
            acc = wmma_bf16(aq1, frag_bT(Ks, LDW, 32, nt * 16), acc);
            sc[nt] = acc;
        }

        // ---- zdot = Q_strip . ksum_prev on the WMMA unit (B = ksum bcast) ----
        v8f zD = {};
        zD = wmma_bf16(aq0, frag_b_bcast(ksbf, 0),  zD);
        zD = wmma_bf16(aq1, frag_b_bcast(ksbf, 32), zD);

        // ---- causal mask + intra rowsum; store scores^T packed (col-major) ----
        float rs[8];
#pragma unroll
        for (int r = 0; r < 8; ++r) rs[r] = 0.0f;
#pragma unroll
        for (int nt = 0; nt < 4; ++nt) {
            const int colt = nt * 16 + n_lo;
            v8bf pack;
#pragma unroll
            for (int r = 0; r < 8; ++r) {
                const int rowt = wave * 16 + r + hi8;
                const float sv = (colt <= rowt) ? sc[nt][r] : 0.0f;
                rs[r] += sv;
                pack[r] = (__bf16)sv;
            }
            // 8 consecutive rows of column colt -> single 16B LDS store
            *(v8bf*)&As[colt * LDW + wave * 16 + hi8] = pack;
        }
#pragma unroll
        for (int r = 0; r < 8; ++r) {
            float sv = rs[r];
            sv += __shfl_xor(sv, 1);
            sv += __shfl_xor(sv, 2);
            sv += __shfl_xor(sv, 4);
            sv += __shfl_xor(sv, 8);
            if (n_lo == 0) zint[wave * 16 + r + hi8] = sv + zD[r];
        }
        __syncthreads();

        // ---- shared fragments: V^T A-tiles and K B-tiles ----
        v16bf av[MTILES][2], bk[2];
#pragma unroll
        for (int mt = 0; mt < MTILES; ++mt) {
            av[mt][0] = frag_a(VsT, LDW, mt * 16, 0);
            av[mt][1] = frag_a(VsT, LDW, mt * 16, 32);
        }
        bk[0] = frag_b(Ks, LDW, 0,  wave * 16);
        bk[1] = frag_b(Ks, LDW, 32, wave * 16);

        // ---- out^T = V^T * scores^T + S^T * Q^T  (lane: 1 row, 16 cols) ----
        v8f outacc[MTILES];
#pragma unroll
        for (int mt = 0; mt < MTILES; ++mt) {
            v8f acc = {};
            acc = wmma_bf16(av[mt][0], frag_b(As, LDW, 0,  wave * 16), acc);
            acc = wmma_bf16(av[mt][1], frag_b(As, LDW, 32, wave * 16), acc);
            acc = wmma_bf16(frag_a(Ss, LDW, mt * 16, 0),  frag_bT(Qs, LDW, 0,  wave * 16), acc);
            acc = wmma_bf16(frag_a(Ss, LDW, mt * 16, 32), frag_bT(Qs, LDW, 32, wave * 16), acc);
            outacc[mt] = acc;
        }

        // ---- state update: S^T += V^T * K (reuses av, bk fragments) ----
#pragma unroll
        for (int mt = 0; mt < MTILES; ++mt) {
            Sacc[mt] = wmma_bf16(av[mt][0], bk[0], Sacc[mt]);
            Sacc[mt] = wmma_bf16(av[mt][1], bk[1], Sacc[mt]);
        }

        // ---- ksum += colsum(phi(K)) via ones^T * K (reuses bk fragments) ----
        {
            v8f cs = {};
            cs = wmma_bf16(ones, bk[0], cs);
            cs = wmma_bf16(ones, bk[1], cs);
            // every row of D equals the column sum; lanes 0..15, VGPR 0 suffice
            if (lane < 16) ksum[wave * 16 + n_lo] += cs[0];
        }

        // ---- normalize and store: out^T lane owns one row, two runs of 8 cons. cols ----
        {
            const int rowt = wave * 16 + n_lo;
            const float rz = 1.0f / (zint[rowt] + 1e-6f);
            float* op = og + (size_t)(t0 + rowt) * rowstride + mo + hi8;
#pragma unroll
            for (int mt = 0; mt < MTILES; ++mt) {
                float4 o0, o1;
                o0.x = outacc[mt][0] * rz; o0.y = outacc[mt][1] * rz;
                o0.z = outacc[mt][2] * rz; o0.w = outacc[mt][3] * rz;
                o1.x = outacc[mt][4] * rz; o1.y = outacc[mt][5] * rz;
                o1.z = outacc[mt][6] * rz; o1.w = outacc[mt][7] * rz;
                *(float4*)(op + mt * 16)     = o0;
                *(float4*)(op + mt * 16 + 4) = o1;
            }
        }
        __syncthreads();  // protect LDS (Ss, ksum, ksbf) before next chunk
    }
}

extern "C" void kernel_launch(void* const* d_in, const int* in_sizes, int n_in,
                              void* d_out, int out_size, void* d_ws, size_t ws_size,
                              hipStream_t stream) {
    const float* q = (const float*)d_in[0];
    const float* k = (const float*)d_in[1];
    const float* v = (const float*)d_in[2];
    float* out = (float*)d_out;
    dim3 grid(8 * HEADS * MSPLIT);   // 256 blocks: (batch, head, m-half)
    dim3 block(128);                 // 4 wave32s
    hipLaunchKernelGGL(LinearAttention_73383811220087_kernel,
                       grid, block, 0, stream, q, k, v, out);
}
